// Qwen2_5_VisionTransformerWrapper_68135361183879
// MI455X (gfx1250) — matrix operand order repaired and checked by reading, not compiled, not run on hardware
//
#include <hip/hip_runtime.h>
#include <hip/hip_bf16.h>
#include <math.h>

// ---------------------------------------------------------------------------
// Qwen2.5-ViT block stack for MI455X (gfx1250, wave32, WMMA).
// Strategy: bf16 WMMA 16x16x32 for all GEMMs + flash attention, f32 accum.
// GEMM uses a 2-deep async global->LDS pipeline (s_wait_asynccnt on the
// previous group only) so HBM latency overlaps WMMA issue.
// Weights converted fp32->bf16 (transposed to [N][K]) each call into a reused
// ~89MB workspace region; activations ~102MB. Total ws need ~192MB.
// ---------------------------------------------------------------------------

typedef __bf16 bf16;
typedef __attribute__((ext_vector_type(16))) __bf16 v16bf;
typedef __attribute__((ext_vector_type(8)))  float  v8f;
typedef __attribute__((ext_vector_type(4)))  unsigned int u32x4;
typedef __attribute__((ext_vector_type(4)))  int v4i;

static constexpr int S    = 2048;
static constexpr int HIDC = 1280;
static constexpr int NH   = 16;
static constexpr int DH   = 80;
static constexpr int DHP  = 96;    // head dim padded to multiple of 32 (WMMA K)
static constexpr int INTC = 3420;
static constexpr int INTP = 3456;  // MLP dim padded to multiple of 128
static constexpr int NL   = 8;
static constexpr int MOUT = 3584;
static constexpr int MH   = 5120;  // merger hidden (4*HID)

union FragAB { u32x4 u[2]; v16bf v; };

__device__ inline v8f vzero8() {
  v8f z;
  #pragma unroll
  for (int i = 0; i < 8; ++i) z[i] = 0.f;
  return z;
}

__device__ inline v8f wmma_bf16(const FragAB& a, const FragAB& b, v8f c) {
  return __builtin_amdgcn_wmma_f32_16x16x32_bf16(false, a.v, false, b.v,
                                                 (short)0, c, false, false);
}

#if __has_builtin(__builtin_amdgcn_global_load_async_to_lds_b128) && \
    __has_builtin(__builtin_amdgcn_s_wait_asynccnt)
#define HAVE_ASYNC_LDS 1
// Builtin expects v4i pointers: AS1 (global) source, AS3 (LDS) destination.
// generic->AS pointer via integer round-trip (flat LDS addr low 32 bits == LDS offset)
#define GLB_V4I(p) ((__attribute__((address_space(1))) v4i*)(unsigned long long)(p))
#define LDS_V4I(p) ((__attribute__((address_space(3))) v4i*)(unsigned int)(unsigned long long)(p))
#endif

// ---------------------------------------------------------------------------
// Weight convert + transpose: W[K][N] f32 -> Wt[Npad][Kpad] bf16, zero padded.
// ---------------------------------------------------------------------------
__global__ void k_convert_wT(const float* __restrict__ W, bf16* __restrict__ Wt,
                             int K, int N, int Kpad, int Npad) {
  long long idx = (long long)blockIdx.x * blockDim.x + threadIdx.x;
  long long tot = (long long)Kpad * Npad;
  if (idx >= tot) return;
  int n = (int)(idx / Kpad);
  int k = (int)(idx % Kpad);
  float v = (n < N && k < K) ? W[(long long)k * N + n] : 0.f;
  Wt[idx] = (bf16)v;
}

// ---------------------------------------------------------------------------
// RMSNorm over C columns, f32 in, bf16 out. One block (256 thr) per row.
// ---------------------------------------------------------------------------
__global__ __launch_bounds__(256)
void k_rmsnorm(const float* __restrict__ x, const float* __restrict__ w,
               bf16* __restrict__ y, int C) {
  int row = blockIdx.x;
  const float* xr = x + (long long)row * C;
  float s = 0.f;
  for (int c = threadIdx.x; c < C; c += 256) { float v = xr[c]; s += v * v; }
  __shared__ float red[256];
  red[threadIdx.x] = s;
  __syncthreads();
  for (int off = 128; off > 0; off >>= 1) {
    if ((int)threadIdx.x < off) red[threadIdx.x] += red[threadIdx.x + off];
    __syncthreads();
  }
  float inv = rsqrtf(red[0] / (float)C + 1e-6f);
  for (int c = threadIdx.x; c < C; c += 256)
    y[(long long)row * C + c] = (bf16)(xr[c] * inv * w[c]);
}

// ---------------------------------------------------------------------------
// Tiled bf16 WMMA GEMM: C[M][N] = A[M][K] * Bt[N][K]^T + bias, epilogues.
// 256 threads = 8 waves (2x4), per-wave 64x32 tile, BK=32.
// Double-buffered LDS; async DMA for tile kt+1 overlaps WMMA on tile kt.
// ---------------------------------------------------------------------------
#define EPI_STORE_BF16 0
#define EPI_RES_F32    1
#define EPI_SILU_F32   2
#define EPI_MUL_BF16   3
#define EPI_GELU_BF16  4
#define EPI_STORE_F32  5

template <int EPI>
__global__ __launch_bounds__(256)
void k_gemm(const bf16* __restrict__ A, const bf16* __restrict__ Bt,
            const float* __restrict__ bias, int M, int N, int K, int nbias,
            float* __restrict__ outF, bf16* __restrict__ outB,
            const float* __restrict__ aux) {
  __shared__ bf16 As[2][128 * 32];
  __shared__ bf16 Bs[2][128 * 32];
  const int tid  = threadIdx.x;
  const int lane = tid & 31, wave = tid >> 5;
  const int wr = wave >> 2, wc = wave & 3;           // 2 (rows) x 4 (cols)
  const int halfsel = lane >> 4, l16 = lane & 15;
  const long long bm = (long long)blockIdx.y * 128;
  const long long bn = (long long)blockIdx.x * 128;
  const int nk = K >> 5;

  // stage one 128x32 tile pair; 512 x 16B chunks per matrix, 2 per thread
  auto stage = [&](int kt, int buf) {
    #pragma unroll
    for (int it = 0; it < 2; ++it) {
      int ch = tid + it * 256;
      int r = ch >> 2, c = (ch & 3) << 3;
      const bf16* ga = A  + (bm + r) * K + kt * 32 + c;
      const bf16* gb = Bt + (bn + r) * K + kt * 32 + c;
#ifdef HAVE_ASYNC_LDS
      __builtin_amdgcn_global_load_async_to_lds_b128(GLB_V4I(ga), LDS_V4I(&As[buf][r * 32 + c]), 0, 0);
      __builtin_amdgcn_global_load_async_to_lds_b128(GLB_V4I(gb), LDS_V4I(&Bs[buf][r * 32 + c]), 0, 0);
#else
      *(u32x4*)&As[buf][r * 32 + c] = *(const u32x4*)ga;
      *(u32x4*)&Bs[buf][r * 32 + c] = *(const u32x4*)gb;
#endif
    }
  };

  v8f acc[4][2];
  #pragma unroll
  for (int mt = 0; mt < 4; ++mt)
    #pragma unroll
    for (int nt = 0; nt < 2; ++nt) acc[mt][nt] = vzero8();

  stage(0, 0);  // prologue: DMA for first tile in flight

  for (int kt = 0; kt < nk; ++kt) {
    const int cur = kt & 1;
    if (kt + 1 < nk) {
      stage(kt + 1, cur ^ 1);  // DMA next tile while we compute this one
      // GL2 prefetch of the streamed weight tile two steps ahead (speculative)
      __builtin_prefetch(Bt + (bn + (tid >> 1)) * K + (kt + 2) * 32, 0, 1);
#ifdef HAVE_ASYNC_LDS
      __builtin_amdgcn_s_wait_asynccnt(4);   // only the 4 ops of group kt
#endif
    } else {
#ifdef HAVE_ASYNC_LDS
      __builtin_amdgcn_s_wait_asynccnt(0);
#endif
    }
    __syncthreads();

    FragAB a[4], b[2];
    #pragma unroll
    for (int mt = 0; mt < 4; ++mt) {      // A frag: lanes<16 K{0..7,16..23}
      int r  = wr * 64 + mt * 16 + l16;
      int c0 = halfsel ? 8 : 0;
      a[mt].u[0] = *(const u32x4*)&As[cur][r * 32 + c0];
      a[mt].u[1] = *(const u32x4*)&As[cur][r * 32 + c0 + 16];
    }
    #pragma unroll
    for (int nt = 0; nt < 2; ++nt) {      // B frag: lanes<16 K{0..7,16..23}
      int r  = wc * 32 + nt * 16 + l16;
      int c0 = halfsel ? 8 : 0;
      b[nt].u[0] = *(const u32x4*)&Bs[cur][r * 32 + c0];
      b[nt].u[1] = *(const u32x4*)&Bs[cur][r * 32 + c0 + 16];
    }
    #pragma unroll
    for (int mt = 0; mt < 4; ++mt)
      #pragma unroll
      for (int nt = 0; nt < 2; ++nt)
        acc[mt][nt] = wmma_bf16(a[mt], b[nt], acc[mt][nt]);
    __syncthreads();  // all waves done reading buf cur before it is re-staged
  }

  // epilogue; C layout: VGPR j -> row j + 8*(lane>=16), col = lane&15
  #pragma unroll
  for (int mt = 0; mt < 4; ++mt) {
    #pragma unroll
    for (int nt = 0; nt < 2; ++nt) {
      #pragma unroll
      for (int j = 0; j < 8; ++j) {
        long long row = bm + wr * 64 + mt * 16 + j + 8 * halfsel;
        long long col = bn + wc * 32 + nt * 16 + l16;
        float v = acc[mt][nt][j];
        if (bias && col < nbias) v += bias[col];
        long long o = row * N + col;
        if      (EPI == EPI_STORE_BF16) outB[o] = (bf16)v;
        else if (EPI == EPI_RES_F32)    outF[o] += v;
        else if (EPI == EPI_SILU_F32)   outF[o] = v / (1.f + __expf(-v));
        else if (EPI == EPI_MUL_BF16)   outB[o] = (bf16)(v * aux[o]);
        else if (EPI == EPI_GELU_BF16)  outB[o] = (bf16)(0.5f * v * (1.f + erff(v * 0.70710678118f)));
        else                            outF[o] = v;
      }
    }
  }
}

// ---------------------------------------------------------------------------
// RoPE + head split. qkv[S][3*HID] bf16 -> qp/kp [H][S][96] (zero padded,
// q pre-scaled by 1/sqrt(D)), vT [H][80][S] (transposed for WMMA B frags).
// ---------------------------------------------------------------------------
__global__ void k_rope_split(const bf16* __restrict__ qkv,
                             const float* __restrict__ cosb, const float* __restrict__ sinb,
                             bf16* __restrict__ qp, bf16* __restrict__ kp,
                             bf16* __restrict__ vt) {
  int s = blockIdx.x, h = blockIdx.y, d = threadIdx.x;   // block = 96 threads
  long long obase = ((long long)h * S + s) * DHP + d;
  if (d >= DH) { qp[obase] = (bf16)0.f; kp[obase] = (bf16)0.f; return; }
  const bf16* rowp = qkv + (long long)s * 3 * HIDC;
  float q  = (float)rowp[h * DH + d];
  float k  = (float)rowp[HIDC + h * DH + d];
  float v  = (float)rowp[2 * HIDC + h * DH + d];
  int   dp = (d < 40) ? d + 40 : d - 40;
  float qr = (float)rowp[h * DH + dp];
  float kr = (float)rowp[HIDC + h * DH + dp];
  float sg = (d < 40) ? -1.f : 1.f;
  float c  = cosb[s * DH + d], sn = sinb[s * DH + d];
  qp[obase] = (bf16)((q * c + sg * qr * sn) * 0.11180339887498948f);  // 1/sqrt(80)
  kp[obase] = (bf16)(k * c + sg * kr * sn);
  vt[((long long)h * DH + d) * S + s] = (bf16)v;
}

// ---------------------------------------------------------------------------
// Flash attention: one block = (64-row q tile, head), 4 waves, wave = 16 rows.
// full=1: loop all 32 kv blocks (layers 3,7); full=0: diagonal block only.
// ---------------------------------------------------------------------------
__global__ __launch_bounds__(128)
void k_attn(const bf16* __restrict__ qp, const bf16* __restrict__ kp,
            const bf16* __restrict__ vt, bf16* __restrict__ out,
            const float* __restrict__ fmaskp, const float* __restrict__ wmaskp,
            int full) {
  __shared__ bf16 P[4][16][64];
  const int qb = blockIdx.x, h = blockIdx.y;
  const int lane = threadIdx.x & 31, w = threadIdx.x >> 5;
  const int halfsel = lane >> 4, l16 = lane & 15;
  const int qr0 = qb * 64 + w * 16;

  // Q fragments (A layout), 3 k-steps over padded 96
  FragAB qa[3];
  {
    const bf16* qrow = qp + ((long long)h * S + qr0 + l16) * DHP;
    #pragma unroll
    for (int ks = 0; ks < 3; ++ks) {
      int c0 = ks * 32 + (halfsel ? 8 : 0);
      qa[ks].u[0] = *(const u32x4*)(qrow + c0);
      qa[ks].u[1] = *(const u32x4*)(qrow + c0 + 16);
    }
  }

  float mst[8], lst[8];
  v8f oacc[5];
  #pragma unroll
  for (int j = 0; j < 8; ++j) { mst[j] = -3.0e38f; lst[j] = 0.f; }
  #pragma unroll
  for (int n = 0; n < 5; ++n) oacc[n] = vzero8();

  const int kb0 = full ? 0 : qb, kb1 = full ? (S / 64) : qb + 1;
  for (int kb = kb0; kb < kb1; ++kb) {
    // S = Q K^T (scale folded into Q)
    v8f sc[4];
    #pragma unroll
    for (int nt = 0; nt < 4; ++nt) {
      v8f z = vzero8();
      const bf16* krow = kp + ((long long)h * S + kb * 64 + nt * 16 + l16) * DHP;
      #pragma unroll
      for (int ks = 0; ks < 3; ++ks) {
        FragAB kf;
        int c0 = ks * 32 + (halfsel ? 8 : 0);
        kf.u[0] = *(const u32x4*)(krow + c0);
        kf.u[1] = *(const u32x4*)(krow + c0 + 16);
        z = wmma_bf16(qa[ks], kf, z);
      }
      sc[nt] = z;
    }
    // additive mask: (1 - m) * FLT_MIN_NEG
    #pragma unroll
    for (int nt = 0; nt < 4; ++nt) {
      #pragma unroll
      for (int j = 0; j < 8; ++j) {
        int rl = w * 16 + j + 8 * halfsel;
        int cl = nt * 16 + l16;
        float mv = full ? fmaskp[(long long)(qb * 64 + rl) * S + kb * 64 + cl]
                        : wmaskp[(long long)qb * 4096 + rl * 64 + cl];
        sc[nt][j] += (1.f - mv) * -3.4028235e38f;
      }
    }
    // online softmax stats (rows live per-VGPR; reduce across 16-lane halves)
    float alpha[8];
    #pragma unroll
    for (int j = 0; j < 8; ++j) {
      float t = fmaxf(fmaxf(sc[0][j], sc[1][j]), fmaxf(sc[2][j], sc[3][j]));
      #pragma unroll
      for (int m = 1; m < 16; m <<= 1) t = fmaxf(t, __shfl_xor(t, m, 32));
      float mn = fmaxf(mst[j], t);
      alpha[j] = __expf(mst[j] - mn);
      mst[j] = mn;
    }
    #pragma unroll
    for (int j = 0; j < 8; ++j) {
      float s0 = 0.f;
      #pragma unroll
      for (int nt = 0; nt < 4; ++nt) {
        float p = __expf(sc[nt][j] - mst[j]);
        sc[nt][j] = p;
        s0 += p;
      }
      #pragma unroll
      for (int m = 1; m < 16; m <<= 1) s0 += __shfl_xor(s0, m, 32);
      lst[j] = lst[j] * alpha[j] + s0;
    }
    #pragma unroll
    for (int n = 0; n < 5; ++n)
      #pragma unroll
      for (int j = 0; j < 8; ++j) oacc[n][j] *= alpha[j];

    // P: C layout -> LDS row-major -> A layout
    __syncthreads();
    #pragma unroll
    for (int nt = 0; nt < 4; ++nt)
      #pragma unroll
      for (int j = 0; j < 8; ++j)
        P[w][j + 8 * halfsel][nt * 16 + l16] = (bf16)sc[nt][j];
    __syncthreads();

    // O += P V  (V rows come from vT[h][d][s], already B-frag friendly)
    #pragma unroll
    for (int ks = 0; ks < 2; ++ks) {
      FragAB pa;
      int c0 = ks * 32 + (halfsel ? 8 : 0);
      pa.u[0] = *(const u32x4*)&P[w][l16][c0];
      pa.u[1] = *(const u32x4*)&P[w][l16][c0 + 16];
      #pragma unroll
      for (int n = 0; n < 5; ++n) {
        const bf16* vrow = vt + ((long long)h * DH + n * 16 + l16) * S + kb * 64;
        FragAB vf;
        int cv = ks * 32 + (halfsel ? 8 : 0);
        vf.u[0] = *(const u32x4*)(vrow + cv);
        vf.u[1] = *(const u32x4*)(vrow + cv + 16);
        oacc[n] = wmma_bf16(pa, vf, oacc[n]);
      }
    }
  }

  // out[s][h*80+d], already in (S, H*D) interleave of the reference
  #pragma unroll
  for (int n = 0; n < 5; ++n)
    #pragma unroll
    for (int j = 0; j < 8; ++j) {
      long long row = qb * 64 + w * 16 + j + 8 * halfsel;
      long long col = h * DH + n * 16 + l16;
      out[row * HIDC + col] = (bf16)(oacc[n][j] / lst[j]);
    }
}

// ---------------------------------------------------------------------------
// Host-side orchestration
// ---------------------------------------------------------------------------
static inline void conv_wT(const float* W, bf16* Wt, int K, int N, int Kp, int Np,
                           hipStream_t st) {
  long long tot = (long long)Kp * Np;
  int grid = (int)((tot + 255) / 256);
  k_convert_wT<<<grid, 256, 0, st>>>(W, Wt, K, N, Kp, Np);
}

extern "C" void kernel_launch(void* const* d_in, const int* in_sizes, int n_in,
                              void* d_out, int out_size, void* d_ws, size_t ws_size,
                              hipStream_t stream) {
  (void)in_sizes; (void)n_in; (void)out_size; (void)ws_size;
  const float* hs_in = (const float*)d_in[0];
  const float* fmask = (const float*)d_in[1];
  const float* wmask = (const float*)d_in[2];
  const float* cosb  = (const float*)d_in[3];
  const float* sinb  = (const float*)d_in[4];
  const float* n1w   = (const float*)d_in[5];
  const float* n2w   = (const float*)d_in[6];
  const float* qkvW  = (const float*)d_in[7];
  const float* qkvB  = (const float*)d_in[8];
  const float* projW = (const float*)d_in[9];
  const float* projB = (const float*)d_in[10];
  const float* gateW = (const float*)d_in[11];
  const float* gateB = (const float*)d_in[12];
  const float* upW   = (const float*)d_in[13];
  const float* upB   = (const float*)d_in[14];
  const float* downW = (const float*)d_in[15];
  const float* downB = (const float*)d_in[16];
  const float* lnq   = (const float*)d_in[17];
  const float* fc1W  = (const float*)d_in[18];
  const float* fc1B  = (const float*)d_in[19];
  const float* fc2W  = (const float*)d_in[20];
  const float* fc2B  = (const float*)d_in[21];

  char* ws = (char*)d_ws;
  size_t off = 0;
  auto take = [&](size_t b) -> char* {
    char* p = ws + off;
    off = (off + b + 255) & ~(size_t)255;
    return p;
  };

  // Reused bf16 weight region: max(per-layer 39.7MB, fc1+fc2 89.2MB)
  bf16* wreg  = (bf16*)take(89128960);
  bf16* qkvT  = wreg;                              // [3840][1280]
  bf16* projT = qkvT + (size_t)3840 * 1280;        // [1280][1280]
  bf16* gateT = projT + (size_t)1280 * 1280;       // [3456][1280] zero-padded
  bf16* upT   = gateT + (size_t)INTP * 1280;       // [3456][1280]
  bf16* downT = upT   + (size_t)INTP * 1280;       // [1280][3456] zero-padded K
  bf16* fc1T  = wreg;                              // [5120][5120] (after layers)
  bf16* fc2T  = wreg + (size_t)MH * MH;            // [3584][5120]

  float* hs   = (float*)take((size_t)S * HIDC * 4);
  bf16*  xn   = (bf16*) take((size_t)S * HIDC * 2);
  bf16*  qkvb = (bf16*) take((size_t)S * 3 * HIDC * 2);
  bf16*  qp   = (bf16*) take((size_t)NH * S * DHP * 2);
  bf16*  kp   = (bf16*) take((size_t)NH * S * DHP * 2);
  bf16*  vt   = (bf16*) take((size_t)NH * DH * S * 2);
  bf16*  att  = (bf16*) take((size_t)S * HIDC * 2);
  float* gbuf = (float*)take((size_t)S * INTP * 4);
  bf16*  hmid = (bf16*) take((size_t)S * INTP * 2);
  bf16*  mbuf = (bf16*) take((size_t)512 * MH * 2);

  (void)hipMemcpyAsync(hs, hs_in, (size_t)S * HIDC * 4, hipMemcpyDeviceToDevice, stream);

  for (int i = 0; i < NL; ++i) {
    // convert this layer's weights (stream-ordered, region reuse is safe)
    conv_wT(qkvW  + (size_t)i * HIDC * 3 * HIDC, qkvT, 1280, 3840, 1280, 3840, stream);
    conv_wT(projW + (size_t)i * HIDC * HIDC,     projT, 1280, 1280, 1280, 1280, stream);
    conv_wT(gateW + (size_t)i * HIDC * INTC,     gateT, 1280, INTC, 1280, INTP, stream);
    conv_wT(upW   + (size_t)i * HIDC * INTC,     upT,   1280, INTC, 1280, INTP, stream);
    conv_wT(downW + (size_t)i * INTC * HIDC,     downT, INTC, 1280, INTP, 1280, stream);

    // attention block
    k_rmsnorm<<<S, 256, 0, stream>>>(hs, n1w + i * HIDC, xn, HIDC);
    k_gemm<EPI_STORE_BF16><<<dim3(3840 / 128, S / 128), 256, 0, stream>>>(
        xn, qkvT, qkvB + (size_t)i * 3840, S, 3840, 1280, 3840, nullptr, qkvb, nullptr);
    k_rope_split<<<dim3(S, NH), DHP, 0, stream>>>(qkvb, cosb, sinb, qp, kp, vt);
    k_attn<<<dim3(S / 64, NH), 128, 0, stream>>>(qp, kp, vt, att, fmask, wmask,
                                                 (i == 3 || i == 7) ? 1 : 0);
    k_gemm<EPI_RES_F32><<<dim3(HIDC / 128, S / 128), 256, 0, stream>>>(
        att, projT, projB + (size_t)i * HIDC, S, HIDC, 1280, HIDC, hs, nullptr, nullptr);

    // MLP block
    k_rmsnorm<<<S, 256, 0, stream>>>(hs, n2w + i * HIDC, xn, HIDC);
    k_gemm<EPI_SILU_F32><<<dim3(INTP / 128, S / 128), 256, 0, stream>>>(
        xn, gateT, gateB + (size_t)i * INTC, S, INTP, 1280, INTC, gbuf, nullptr, nullptr);
    k_gemm<EPI_MUL_BF16><<<dim3(INTP / 128, S / 128), 256, 0, stream>>>(
        xn, upT, upB + (size_t)i * INTC, S, INTP, 1280, INTC, nullptr, hmid, gbuf);
    k_gemm<EPI_RES_F32><<<dim3(HIDC / 128, S / 128), 256, 0, stream>>>(
        hmid, downT, downB + (size_t)i * HIDC, S, HIDC, INTP, HIDC, hs, nullptr, nullptr);
  }

  // patch merger: RMSNorm -> view [512][5120] -> fc1+GELU -> fc2
  k_rmsnorm<<<S, 256, 0, stream>>>(hs, lnq, xn, HIDC);
  conv_wT(fc1W, fc1T, MH, MH, MH, MH, stream);
  conv_wT(fc2W, fc2T, MH, MOUT, MH, MOUT, stream);
  k_gemm<EPI_GELU_BF16><<<dim3(MH / 128, 512 / 128), 256, 0, stream>>>(
      xn, fc1T, fc1B, 512, MH, MH, MH, nullptr, mbuf, nullptr);
  k_gemm<EPI_STORE_F32><<<dim3(MOUT / 128, 512 / 128), 256, 0, stream>>>(
      mbuf, fc2T, fc2B, 512, MOUT, MH, MOUT, (float*)d_out, nullptr, nullptr);
}
